// Conv2dKan_69234872812237
// MI455X (gfx1250) — matrix operand order repaired and checked by reading, not compiled, not run on hardware
//
#include <hip/hip_runtime.h>

// ---------------------------------------------------------------------------
// KAN 3x3 conv as implicit GEMM on CDNA5 WMMA (bf16 -> f32 accumulate).
//   out[b,o,l] = sum_{i,k,a} W'[k][o][i*9+a] * f_a(x[b,i, l+k])
//   f_0..7 = Hermite H_0..H_7,  f_8 = silu.
// M=128 (Cout), N=16384 (B*H*W), K=5184 (576 feat-ch * 9 taps).
//
// Block = 8 waves; wave mw owns Cout rows [mw*16, mw*16+16) and BOTH
// 16-pixel halves of one image row -> each A tile feeds two WMMAs
// (1 global b128 pair per 2 wmma instead of per 1).
// ---------------------------------------------------------------------------

#define BATCH   16
#define CIN     64
#define COUT    128
#define HH      32
#define WW      32
#define NFEAT   9            // 8 Hermite + silu per (channel, value)
#define CF      (CIN * NFEAT)      // 576 feature channels
#define CIN_CHUNK 32
#define CF_CHUNK  (CIN_CHUNK * NFEAT)   // 288 = 9 K-chunks of 32
#define CF_PAD    296                   // pad LDS cf-stride for bank spread
#define HALO_W    (WW + 2)              // 34

typedef __attribute__((ext_vector_type(16))) __bf16 v16bf;
typedef __attribute__((ext_vector_type(8)))  float  v8f;

union BfVec {
    v16bf v;
    uint4 q[2];
};

static __device__ __forceinline__ unsigned short f2bf(float f) {
    return __builtin_bit_cast(unsigned short, (__bf16)f);
}

// Hermite H_0..H_7 (physicists') + silu residual.
static __device__ __forceinline__ void kan_features(float x, float f[NFEAT]) {
    float hm1 = 1.0f;          // H_0
    float h   = 2.0f * x;      // H_1
    f[0] = hm1;
    f[1] = h;
#pragma unroll
    for (int n = 1; n < NFEAT - 2; ++n) {   // H_2 .. H_7
        float hn = 2.0f * x * h - 2.0f * (float)n * hm1;
        f[n + 1] = hn;
        hm1 = h;
        h = hn;
    }
    f[8] = x / (1.0f + __expf(-x));         // silu
}

// ---------------------------------------------------------------------------
// Prologue: fold w_s*c and w_b into bf16 W'[tap][o][cf], cf = i*9 + a.
// ---------------------------------------------------------------------------
__global__ void kan_build_weights(const float* __restrict__ w_b,
                                  const float* __restrict__ w_s,
                                  const float* __restrict__ c,
                                  unsigned short* __restrict__ Wp,
                                  int n) {
    int idx = blockIdx.x * blockDim.x + threadIdx.x;
    if (idx >= n) return;
    int cf  = idx % CF;
    int o   = (idx / CF) % COUT;
    int tap = idx / (CF * COUT);
    int i   = cf / NFEAT;
    int a   = cf % NFEAT;
    size_t base = ((size_t)i * COUT + o) * 9 + tap;   // [Cin,Cout,9,1] layout
    float v = (a < 8) ? w_s[base] * c[base * 8 + a]   // spline: w_s * c
                      : w_b[base];                    // residual: w_b
    Wp[idx] = f2bf(v);
}

// ---------------------------------------------------------------------------
// Main: block = 256 threads = 8 Cout-waves; block covers 128 Cout x 32 px
// (one full image row of one batch image).
// ---------------------------------------------------------------------------
__global__ __launch_bounds__(256)
void kan_conv_wmma(const float* __restrict__ x,
                   const unsigned short* __restrict__ Wp,
                   float* __restrict__ out) {
    // bf16 features, cf-fastest so a K-half of 16 values is one contiguous
    // 32-byte run -> two ds_load_b128 per B operand per lane.
    __shared__ __align__(16) unsigned short feat[3][HALO_W][CF_PAD];

    const int tid  = threadIdx.x;
    const int lane = tid & 31;
    const int mw   = tid >> 5;          // Cout sub-tile (8 x 16 = 128)

    const int b  = blockIdx.x / HH;
    const int oy = blockIdx.x % HH;
    const int o_row0 = mw * 16;

    const int m    = lane & 15;         // WMMA A row / B column within tile
    const int half = lane >> 4;         // K-halves per 16-bit A/B striping

    v8f acc0 = {};                      // pixels [0,16)
    v8f acc1 = {};                      // pixels [16,32)

    for (int chunk = 0; chunk < CIN / CIN_CHUNK; ++chunk) {
        // ---- Phase 1: stage features for 32 channels x 3 rows x 34 cols ----
        const int ci0 = chunk * CIN_CHUNK;
        for (int idx = tid; idx < CIN_CHUNK * 3 * HALO_W; idx += 256) {
            int ci  = idx / (3 * HALO_W);
            int rem = idx % (3 * HALO_W);
            int row = rem / HALO_W;     // 0..2  -> iy = oy + row - 1
            int col = rem % HALO_W;     // 0..33 -> ix = col - 1
            int iy = oy + row - 1;
            int ix = col - 1;
            // Zero-padded x: padded lanes still get full features (H_a(0)!=0).
            float xv = 0.0f;
            if (iy >= 0 && iy < HH && ix >= 0 && ix < WW)
                xv = x[(((size_t)b * CIN + ci0 + ci) * HH + iy) * WW + ix];
            float f[NFEAT];
            kan_features(xv, f);
            unsigned short* dst = &feat[row][col][ci * NFEAT];
#pragma unroll
            for (int a = 0; a < NFEAT; ++a) dst[a] = f2bf(f[a]);
        }
        __syncthreads();

        // ---- Phase 2: 9 taps x 9 K-chunks of 32; A loaded once per pair of
        //      v_wmma_f32_16x16x32_bf16 (two pixel halves share A) ----------
        for (int tap = 0; tap < 9; ++tap) {
            const int dy = tap / 3;
            const int dx = tap % 3;
            const unsigned short* arow =
                Wp + ((size_t)(tap * COUT + o_row0 + m)) * CF
                   + chunk * CF_CHUNK + half * 16;         // A: Cout row
            const unsigned short* brow0 = &feat[dy][m + dx][half * 16];
            const unsigned short* brow1 = &feat[dy][16 + m + dx][half * 16];
            // Speculative prefetch of next tap's weight row into cache.
            if (tap < 8)
                __builtin_prefetch(arow + COUT * CF, 0, 0);
#pragma unroll
            for (int kc = 0; kc < CF_CHUNK / 32; ++kc) {
                // A 16x32 bf16: lane holds K [h*16, h*16+16).
                BfVec a;
                const uint4* ap = (const uint4*)(arow + kc * 32);
                a.q[0] = ap[0];       // K half*16 + 0..7  -> VGPR 0..3
                a.q[1] = ap[1];       // K half*16 + 8..15 -> VGPR 4..7
                // B 32x16 bf16: lane n holds K [half*16, half*16+16).
                BfVec b0, b1;
                const uint4* bp0 = (const uint4*)(brow0 + kc * 32);
                const uint4* bp1 = (const uint4*)(brow1 + kc * 32);
                b0.q[0] = bp0[0];
                b0.q[1] = bp0[1];
                b1.q[0] = bp1[0];
                b1.q[1] = bp1[1];
                acc0 = __builtin_amdgcn_wmma_f32_16x16x32_bf16(
                    false, a.v, false, b0.v, (short)0, acc0, false, false);
                acc1 = __builtin_amdgcn_wmma_f32_16x16x32_bf16(
                    false, a.v, false, b1.v, (short)0, acc1, false, false);
            }
        }
        __syncthreads();
    }

    // ---- Epilogue: 16x16 f32 C/D layout -> out[b, o, oy, ox] ----
#pragma unroll
    for (int r = 0; r < 8; ++r) {
        int o = o_row0 + half * 8 + r;
        size_t base = (((size_t)b * COUT + o) * HH + oy) * WW;
        out[base + m]      = acc0[r];
        out[base + 16 + m] = acc1[r];
    }
}

extern "C" void kernel_launch(void* const* d_in, const int* in_sizes, int n_in,
                              void* d_out, int out_size, void* d_ws, size_t ws_size,
                              hipStream_t stream) {
    const float* x   = (const float*)d_in[0];   // [16,64,32,32]
    const float* w_b = (const float*)d_in[1];   // [64,128,9,1]
    const float* w_s = (const float*)d_in[2];   // [64,128,9,1]
    const float* c   = (const float*)d_in[3];   // [64,128,9,1,8]

    // Workspace: W'[9][128][576] bf16 = 1.27 MB.
    unsigned short* Wp = (unsigned short*)d_ws;
    const int nW = 9 * COUT * CF;
    kan_build_weights<<<(nW + 255) / 256, 256, 0, stream>>>(w_b, w_s, c, Wp, nW);

    dim3 grid(BATCH * HH);              // 512 blocks x 8 waves: full 128 Cout
    kan_conv_wmma<<<grid, 256, 0, stream>>>(x, Wp, (float*)d_out);
}
